// LocalAttentionLayer_45818711113840
// MI455X (gfx1250) — matrix operand order repaired and checked by reading, not compiled, not run on hardware
//
#include <hip/hip_runtime.h>
#include <hip/hip_bf16.h>
#include <stdint.h>

// Problem sizes (fixed by reference)
#define B_  4
#define N_  2048
#define H_  8
#define D_  64
#define DM_ 512
#define M_  (B_ * N_)   // 8192 rows for the projection GEMMs

typedef __attribute__((ext_vector_type(16))) __bf16 bf16x16;
typedef __attribute__((ext_vector_type(8)))  __bf16 bf16x8;
typedef __attribute__((ext_vector_type(4)))  __bf16 bf16x4;
typedef __attribute__((ext_vector_type(8)))  float  f32x8;
typedef __attribute__((ext_vector_type(4)))  float  f32x4;
typedef __attribute__((ext_vector_type(4)))  unsigned int u32x4;
typedef __attribute__((ext_vector_type(8)))  int    i32x8;
typedef __attribute__((ext_vector_type(4)))  int    i32x4;

#if __has_builtin(__builtin_amdgcn_tensor_load_to_lds)
#define USE_TDM 1
#else
#define USE_TDM 0
#endif

// ---------------------------------------------------------------------------
// WMMA helpers
// ---------------------------------------------------------------------------
__device__ __forceinline__ f32x8 wmma_bf16(bf16x16 a, bf16x16 b, f32x8 c) {
  // (neg_a, A, neg_b, B, c_mod, C, reuse_a, reuse_b)
  return __builtin_amdgcn_wmma_f32_16x16x32_bf16(false, a, false, b, (short)0, c,
                                                 false, false);
}

// A-fragment (16x32 bf16): lane m = lane&15 is the row; half = lane>>4 selects
// K sub-chunks. Elements 0..7  <-> K = kbase + half*8 + 0..7
//              Elements 8..15 <-> K = kbase + 16 + half*8 + 0..7
__device__ __forceinline__ bf16x16 load_a_frag(const __bf16* __restrict__ row,
                                               int half, int kbase) {
  bf16x8 lo = *(const bf16x8*)(row + kbase + half * 8);
  bf16x8 hi = *(const bf16x8*)(row + kbase + 16 + half * 8);
  bf16x16 a;
#pragma unroll
  for (int i = 0; i < 8; ++i) { a[i] = lo[i]; a[8 + i] = hi[i]; }
  return a;
}

// B-fragment (32x16 bf16): lane n = lane&15 is the column; half = lane>>4.
// Elements 0..7 <-> K = kbase + half*8 + 0..7; 8..15 <-> K = kbase + 16 + half*8 + 0..7.
__device__ __forceinline__ bf16x16 load_b_frag(const __bf16* __restrict__ colrow,
                                               int half, int kbase) {
  bf16x8 lo = *(const bf16x8*)(colrow + kbase + half * 8);
  bf16x8 hi = *(const bf16x8*)(colrow + kbase + 16 + half * 8);
  bf16x16 b;
#pragma unroll
  for (int i = 0; i < 8; ++i) { b[i] = lo[i]; b[8 + i] = hi[i]; }
  return b;
}

// ---------------------------------------------------------------------------
// Tensor Data Mover: 2D tile load global -> LDS (bf16 elements, data_size=2B)
// D# built per CDNA5 ISA; groups 2-3 zero (2D).
// This toolchain exposes the 6-arg builtin:
//   (u32x4 g0, i32x8 g1, i32x4 g2, i32x4 g3, i32x8 extra, i32 cpol)
// ---------------------------------------------------------------------------
#if USE_TDM
__device__ __forceinline__ void tdm_load_2d(uint32_t lds_byte_addr,
                                            uint64_t gaddr,
                                            uint32_t tensor_d0,
                                            uint32_t tensor_d1,
                                            uint32_t tile_d0, uint32_t tile_d1,
                                            uint32_t stride0_elems) {
  u32x4 g0;
  g0[0] = 1u;                                  // count=1, user mode, no gather
  g0[1] = lds_byte_addr;                       // lds_addr
  g0[2] = (uint32_t)(gaddr & 0xFFFFFFFFu);     // global_addr[31:0]
  g0[3] = (uint32_t)((gaddr >> 32) & 0x01FFFFFFu) | (2u << 30);  // type=2
  i32x8 g1;
  g1[0] = (int)(1u << 16);                     // data_size=1 -> 2 bytes
  g1[1] = (int)((tensor_d0 & 0xFFFFu) << 16); // tensor_dim0[15:0]
  g1[2] = (int)((tensor_d0 >> 16) | ((tensor_d1 & 0xFFFFu) << 16));
  g1[3] = (int)((tensor_d1 >> 16) | (tile_d0 << 16));  // tile_dim0
  g1[4] = (int)tile_d1;                        // tile_dim1 (tile_dim2 = 0)
  g1[5] = (int)stride0_elems;                  // tensor_dim0_stride[31:0]
  g1[6] = 0;
  g1[7] = 0;
  i32x4 gz = {0, 0, 0, 0};
  i32x8 gz8 = {0, 0, 0, 0, 0, 0, 0, 0};
  __builtin_amdgcn_tensor_load_to_lds(g0, g1, gz, gz, gz8, 0);
}
#endif

template <int NCNT>
__device__ __forceinline__ void wait_tensorcnt() {
#if __has_builtin(__builtin_amdgcn_s_wait_tensorcnt)
  __builtin_amdgcn_s_wait_tensorcnt(NCNT);
#else
  asm volatile("s_wait_tensorcnt %0" ::"i"(NCNT) : "memory");
#endif
}

// ---------------------------------------------------------------------------
// Kernel 1: fp32 -> bf16 conversion of x (vectorized)
// ---------------------------------------------------------------------------
__global__ __launch_bounds__(256) void cvt_x_kernel(const float* __restrict__ x,
                                                    __bf16* __restrict__ xb,
                                                    int n4) {
  int i = blockIdx.x * 256 + threadIdx.x;
  if (i >= n4) return;
  f32x4 v = ((const f32x4*)x)[i];
  bf16x4 o;
#pragma unroll
  for (int j = 0; j < 4; ++j) o[j] = (__bf16)v[j];
  ((bf16x4*)xb)[i] = o;
}

// ---------------------------------------------------------------------------
// Kernel 2: transpose + convert all four weight matrices: Wt[n][k] = W[k][n]
// ---------------------------------------------------------------------------
__global__ __launch_bounds__(256) void cvt_w_kernel(
    const float* __restrict__ Wq, const float* __restrict__ Wk,
    const float* __restrict__ Wv, const float* __restrict__ Wo,
    __bf16* __restrict__ WqT, __bf16* __restrict__ WkT,
    __bf16* __restrict__ WvT, __bf16* __restrict__ WoT) {
  const float* src;
  __bf16* dst;
  switch (blockIdx.y) {
    case 0: src = Wq; dst = WqT; break;
    case 1: src = Wk; dst = WkT; break;
    case 2: src = Wv; dst = WvT; break;
    default: src = Wo; dst = WoT; break;
  }
  int i = blockIdx.x * 256 + threadIdx.x;   // 0 .. 512*512-1
  int k = i >> 9;                           // row of W  (coalesced read)
  int n = i & 511;                          // col of W
  dst[(size_t)n * DM_ + k] = (__bf16)src[i];
}

// ---------------------------------------------------------------------------
// Kernel 3: projection GEMM  C[M_ x 512] = Xb[M_ x 512] * W   (W given as Wt)
// Each wave computes a 16(M) x 64(N) tile; block = 4 waves stacked in M.
// mode 0: store as [b][h][n][d]  (Q and K)
// mode 1: store transposed as [b][h][d][n]  (V)
// ---------------------------------------------------------------------------
__global__ __launch_bounds__(128) void proj_gemm_kernel(
    const __bf16* __restrict__ A, const __bf16* __restrict__ WT,
    __bf16* __restrict__ out, int mode) {
  const int lane = threadIdx.x & 31;
  const int wave = threadIdx.x >> 5;  // 0..3
  const int n16 = lane & 15;
  const int half = lane >> 4;
  const int mBase = blockIdx.x * 64 + wave * 16;
  const int colBase = blockIdx.y * 64;

  f32x8 acc[4] = {};
  const __bf16* arow = A + (size_t)(mBase + n16) * DM_;

  for (int k0 = 0; k0 < DM_; k0 += 32) {
    bf16x16 af = load_a_frag(arow, half, k0);
#pragma unroll
    for (int s = 0; s < 4; ++s) {
      const __bf16* wcol = WT + (size_t)(colBase + s * 16 + n16) * DM_;
      bf16x16 bf = load_b_frag(wcol, half, k0);
      acc[s] = wmma_bf16(af, bf, acc[s]);
    }
  }

#pragma unroll
  for (int s = 0; s < 4; ++s) {
    int col = colBase + s * 16 + n16;
    int h = col >> 6, d = col & 63;
#pragma unroll
    for (int r = 0; r < 8; ++r) {
      int m = mBase + r + 8 * half;
      int b = m >> 11, nI = m & (N_ - 1);
      size_t addr;
      if (mode == 0)
        addr = ((size_t)((b * H_ + h) * N_ + nI)) * 64 + d;      // [b][h][n][d]
      else
        addr = ((size_t)((b * H_ + h) * 64 + d)) * N_ + nI;      // [b][h][d][n]
      out[addr] = (__bf16)acc[s][r];
    }
  }
}

// ---------------------------------------------------------------------------
// Kernel 4: flash attention.  S^T = K * Q^T so the C-layout of P^T matches
// the A-fragment layout for the P*V WMMA (keys processed 32/step).
// Block = 8 waves over one (b, h); each wave owns 16 query rows (128 total).
// K/V 32-key tiles are staged once per block into LDS via the Tensor Data
// Mover (double buffered, TENSORcnt-synchronized); fallback: cooperative
// global->LDS staging.  grid = (N/128, H, B).
//
// LDS layout (manual suballocation of one byte array, base offset 0):
//   [0,      8192)  : Kbuf[2][32][64] bf16   (4KB per buffer)
//   [8192,  16384)  : Vbuf[2][64][32] bf16   (4KB per buffer)
//   [16384, 24576)  : kpen[2048] float       (additive key-mask penalty)
// ---------------------------------------------------------------------------
#define KOFF(buf) ((buf) * 4096)
#define VOFF(buf) (8192 + (buf) * 4096)
#define PENOFF    16384

__global__ __launch_bounds__(256) void attn_kernel(
    const __bf16* __restrict__ Q, const __bf16* __restrict__ K,
    const __bf16* __restrict__ Vt, const uint8_t* __restrict__ mask_k,
    const uint8_t* __restrict__ mask_q, __bf16* __restrict__ att) {
  __shared__ __attribute__((aligned(128))) uint8_t smem[24576];

  const int b = blockIdx.z;
  const int h = blockIdx.y;
  const int tid = threadIdx.x;
  const int lane = tid & 31;
  const int wave = tid >> 5;  // 0..7
  const int n16 = lane & 15;  // query column within tile
  const int half = lane >> 4;
  const int q0 = blockIdx.x * 128 + wave * 16;

  float* kpen = (float*)(smem + PENOFF);
  for (int i = tid; i < N_; i += 256)
    kpen[i] = mask_k[b * N_ + i] ? 0.0f : -1e30f;

  const size_t bh = (size_t)(b * H_ + h);
  const __bf16* qbase = Q + (bh * N_ + q0) * 64;
  const __bf16* kbase = K + bh * N_ * 64;    // [n][d], row stride 64
  const __bf16* vbase = Vt + bh * 64 * N_;   // [d][n], row stride N_

  // Loop-invariant Q^T B-fragments: column n16 = query, K-dim = d (0..63)
  const __bf16* qr = qbase + (size_t)n16 * 64;
  bf16x16 qb0 = load_b_frag(qr, half, 0);
  bf16x16 qb1 = load_b_frag(qr, half, 32);

  const bool qm = mask_q[b * N_ + q0 + n16] != 0;
  const float scale = 0.125f;  // D^-0.5

  f32x8 o[4] = {};
  float mrun = -1e30f, lrun = 0.0f;

  // ---- staging helper ------------------------------------------------------
#if USE_TDM
  const bool issuer = (wave == 0);
#define STAGE(bufi, j0s)                                                      \
  do {                                                                        \
    if (issuer) {                                                             \
      tdm_load_2d(KOFF(bufi), (uint64_t)(uintptr_t)(kbase + (size_t)(j0s) * 64), \
                  64u, (uint32_t)N_, 64u, 32u, 64u);                          \
      tdm_load_2d(VOFF(bufi), (uint64_t)(uintptr_t)(vbase + (size_t)(j0s)),   \
                  (uint32_t)N_, 64u, 32u, 64u, (uint32_t)N_);                 \
    }                                                                         \
  } while (0)
#else
#define STAGE(bufi, j0s)                                                      \
  do {                                                                        \
    {                                                                         \
      int kk = tid >> 3, c = tid & 7; /* 32 rows x 8 chunks */                \
      bf16x8 kv = *(const bf16x8*)(kbase + (size_t)((j0s) + kk) * 64 + c * 8); \
      *(bf16x8*)(smem + KOFF(bufi) + (kk * 64 + c * 8) * 2) = kv;             \
      int d = tid >> 2, c2 = tid & 3; /* 64 rows x 4 chunks */                \
      bf16x8 vv = *(const bf16x8*)(vbase + (size_t)d * N_ + (j0s) + c2 * 8);  \
      *(bf16x8*)(smem + VOFF(bufi) + (d * 32 + c2 * 8) * 2) = vv;             \
    }                                                                         \
  } while (0)
#endif

  STAGE(0, 0);  // prologue

  const int T = N_ / 32;  // 64 key-tile iterations
  for (int it = 0; it < T; ++it) {
    const int cur = it & 1;
    const int j0 = it * 32;
    if (it + 1 < T) STAGE(cur ^ 1, j0 + 32);
#if USE_TDM
    if (issuer) {
      if (it + 1 < T) wait_tensorcnt<2>();  // current buffer's 2 TDM ops done
      else            wait_tensorcnt<0>();
    }
#endif
    __syncthreads();  // staged data (and kpen on it==0) visible to all waves

    const __bf16* Kb = (const __bf16*)(smem + KOFF(cur));
    const __bf16* Vb = (const __bf16*)(smem + VOFF(cur));

    // K A-fragments for key tiles [j0, j0+16) and [j0+16, j0+32)  (from LDS)
    bf16x16 a00 = load_a_frag(Kb + n16 * 64, half, 0);
    bf16x16 a01 = load_a_frag(Kb + n16 * 64, half, 32);
    bf16x16 a10 = load_a_frag(Kb + (16 + n16) * 64, half, 0);
    bf16x16 a11 = load_a_frag(Kb + (16 + n16) * 64, half, 32);

    f32x8 s0 = {}, s1 = {};
    s0 = wmma_bf16(a00, qb0, s0);
    s0 = wmma_bf16(a01, qb1, s0);
    s1 = wmma_bf16(a10, qb0, s1);
    s1 = wmma_bf16(a11, qb1, s1);

    // scale + masks; S^T layout: lane column = query n16, row kk = r+8*half
    float p0[8], p1[8];
    float tmax = -1e30f;
#pragma unroll
    for (int r = 0; r < 8; ++r) {
      int kk = j0 + r + 8 * half;
      float v0 = qm ? (s0[r] * scale + kpen[kk]) : 0.0f;
      float v1 = qm ? (s1[r] * scale + kpen[kk + 16]) : 0.0f;
      p0[r] = v0;
      p1[r] = v1;
      tmax = fmaxf(tmax, fmaxf(v0, v1));
    }
    tmax = fmaxf(tmax, __shfl_xor(tmax, 16, 32));
    float newm = fmaxf(mrun, tmax);
    float alpha = __expf(mrun - newm);
    float rsum = 0.0f;
#pragma unroll
    for (int r = 0; r < 8; ++r) {
      p0[r] = __expf(p0[r] - newm);
      p1[r] = __expf(p1[r] - newm);
      rsum += p0[r] + p1[r];
    }
    rsum += __shfl_xor(rsum, 16, 32);
    lrun = lrun * alpha + rsum;
    mrun = newm;

    // Rescale O accumulators: alpha for query row q lives in lane q (and q+16)
#pragma unroll
    for (int r = 0; r < 8; ++r) {
      float ar = __shfl(alpha, r + 8 * half, 32);
#pragma unroll
      for (int s = 0; s < 4; ++s) o[s][r] *= ar;
    }

    // P^T C-layout == P A-fragment layout (tile0 -> elems 0..7, tile1 -> 8..15)
    bf16x16 pa;
#pragma unroll
    for (int r = 0; r < 8; ++r) {
      pa[r] = (__bf16)p0[r];
      pa[8 + r] = (__bf16)p1[r];
    }

    // O[16q x 64d] += P[16q x 32kk] * V[32kk x 64d]   (V from LDS)
#pragma unroll
    for (int s = 0; s < 4; ++s) {
      const __bf16* vr = Vb + (size_t)(s * 16 + n16) * 32;  // row = d column
      bf16x16 vb = load_b_frag(vr, half, 0);
      o[s] = wmma_bf16(pa, vb, o[s]);
    }

    __syncthreads();  // all waves done reading buf[cur] before it is restaged
  }

  // Normalize and store attention output as bf16 [b][n][h*64+d]
  const size_t orow0 = ((size_t)b * N_ + q0) * DM_ + h * 64;
#pragma unroll
  for (int r = 0; r < 8; ++r) {
    float lr = __shfl(lrun, r + 8 * half, 32);
    float inv = (lr > 0.0f) ? (1.0f / lr) : 0.0f;
#pragma unroll
    for (int s = 0; s < 4; ++s) {
      att[orow0 + (size_t)(r + 8 * half) * DM_ + s * 16 + n16] =
          (__bf16)(o[s][r] * inv);
    }
  }
}

// ---------------------------------------------------------------------------
// Kernel 5: output projection  out[M_ x 512] = att * Wo + bo  (f32 output)
// ---------------------------------------------------------------------------
__global__ __launch_bounds__(128) void out_gemm_kernel(
    const __bf16* __restrict__ A, const __bf16* __restrict__ WT,
    const float* __restrict__ bias, float* __restrict__ out) {
  const int lane = threadIdx.x & 31;
  const int wave = threadIdx.x >> 5;
  const int n16 = lane & 15;
  const int half = lane >> 4;
  const int mBase = blockIdx.x * 64 + wave * 16;
  const int colBase = blockIdx.y * 64;

  f32x8 acc[4] = {};
  const __bf16* arow = A + (size_t)(mBase + n16) * DM_;

  for (int k0 = 0; k0 < DM_; k0 += 32) {
    bf16x16 af = load_a_frag(arow, half, k0);
#pragma unroll
    for (int s = 0; s < 4; ++s) {
      const __bf16* wcol = WT + (size_t)(colBase + s * 16 + n16) * DM_;
      bf16x16 bf = load_b_frag(wcol, half, k0);
      acc[s] = wmma_bf16(af, bf, acc[s]);
    }
  }

#pragma unroll
  for (int s = 0; s < 4; ++s) {
    int col = colBase + s * 16 + n16;
    float bb = bias[col];
#pragma unroll
    for (int r = 0; r < 8; ++r) {
      int m = mBase + r + 8 * half;
      out[(size_t)m * DM_ + col] = acc[s][r] + bb;
    }
  }
}

// ---------------------------------------------------------------------------
// Host-side launcher
// ---------------------------------------------------------------------------
extern "C" void kernel_launch(void* const* d_in, const int* in_sizes, int n_in,
                              void* d_out, int out_size, void* d_ws,
                              size_t ws_size, hipStream_t stream) {
  const float* x = (const float*)d_in[0];
  const float* Wq = (const float*)d_in[1];
  const float* Wk = (const float*)d_in[2];
  const float* Wv = (const float*)d_in[3];
  const float* Wo = (const float*)d_in[4];
  const float* bo = (const float*)d_in[5];
  const uint8_t* mask_k = (const uint8_t*)d_in[6];  // numpy bool = 1 byte
  const uint8_t* mask_q = (const uint8_t*)d_in[7];
  float* out = (float*)d_out;

  char* ws = (char*)d_ws;
  size_t off = 0;
  auto alloc = [&](size_t bytes) -> char* {
    char* p = ws + off;
    off += (bytes + 255) & ~(size_t)255;
    return p;
  };
  __bf16* xb  = (__bf16*)alloc((size_t)M_ * DM_ * 2);       // 8 MiB
  __bf16* WqT = (__bf16*)alloc((size_t)DM_ * DM_ * 2);
  __bf16* WkT = (__bf16*)alloc((size_t)DM_ * DM_ * 2);
  __bf16* WvT = (__bf16*)alloc((size_t)DM_ * DM_ * 2);
  __bf16* WoT = (__bf16*)alloc((size_t)DM_ * DM_ * 2);
  __bf16* Qh  = (__bf16*)alloc((size_t)B_ * H_ * N_ * 64 * 2);  // [b][h][n][d]
  __bf16* Kh  = (__bf16*)alloc((size_t)B_ * H_ * N_ * 64 * 2);  // [b][h][n][d]
  __bf16* Vt  = (__bf16*)alloc((size_t)B_ * H_ * 64 * N_ * 2);  // [b][h][d][n]
  __bf16* att = (__bf16*)alloc((size_t)M_ * DM_ * 2);           // [b][n][h*d]

  // 1) convert x to bf16
  cvt_x_kernel<<<(M_ * DM_ / 4 + 255) / 256, 256, 0, stream>>>(x, xb,
                                                               M_ * DM_ / 4);
  // 2) transpose-convert weights
  cvt_w_kernel<<<dim3(DM_ * DM_ / 256, 4), 256, 0, stream>>>(
      Wq, Wk, Wv, Wo, WqT, WkT, WvT, WoT);
  // 3) Q, K, V projections (WMMA)
  proj_gemm_kernel<<<dim3(M_ / 64, DM_ / 64), 128, 0, stream>>>(xb, WqT, Qh, 0);
  proj_gemm_kernel<<<dim3(M_ / 64, DM_ / 64), 128, 0, stream>>>(xb, WkT, Kh, 0);
  proj_gemm_kernel<<<dim3(M_ / 64, DM_ / 64), 128, 0, stream>>>(xb, WvT, Vt, 1);
  // 4) flash attention (WMMA + TDM-staged K/V tiles in LDS)
  attn_kernel<<<dim3(N_ / 128, H_, B_), 256, 0, stream>>>(Qh, Kh, Vt, mask_k,
                                                          mask_q, att);
  // 5) output projection + bias (WMMA, f32 out)
  out_gemm_kernel<<<dim3(M_ / 64, DM_ / 64), 128, 0, stream>>>(att, WoT, bo,
                                                               out);
}
